// ContextCrossAttnBlock_18786186953488
// MI455X (gfx1250) — matrix operand order repaired and checked by reading, not compiled, not run on hardware
//
#include <hip/hip_runtime.h>
#include <hip/hip_bf16.h>

// ---------------------------------------------------------------------------
// ContextCrossAttnBlock forward for MI455X (gfx1250, wave32, WMMA)
// B=8, C=128, H=W=32, L=1024, FORE=64, POST=256, HEADS=8, hd=16, HID=512
// ---------------------------------------------------------------------------

typedef __attribute__((ext_vector_type(16))) _Float16 v16h;
typedef __attribute__((ext_vector_type(8)))  float    v8f;

#define NB   8
#define NC   128
#define NL   1024
#define NH   32
#define NW   32
#define NHID 512
#define NHEADS 8
#define NHD  16

// ---------------------------------------------------------------------------
// WMMA fragment builders (CDNA5 16x16x32 f16 layouts, wave32).
// A (16x32 slice, M = lane%16): two contiguous 8-float runs per lane.
// B (32x16 slice, N = lane%16): two contiguous 8-float runs per lane.
// All loads unconditional float4 (global_load_b128), 16B aligned.
// ---------------------------------------------------------------------------
__device__ __forceinline__ v16h cvt16(float4 a, float4 b, float4 c, float4 d) {
  v16h r;
  r[0]  = (_Float16)a.x; r[1]  = (_Float16)a.y; r[2]  = (_Float16)a.z; r[3]  = (_Float16)a.w;
  r[4]  = (_Float16)b.x; r[5]  = (_Float16)b.y; r[6]  = (_Float16)b.z; r[7]  = (_Float16)b.w;
  r[8]  = (_Float16)c.x; r[9]  = (_Float16)c.y; r[10] = (_Float16)c.z; r[11] = (_Float16)c.w;
  r[12] = (_Float16)d.x; r[13] = (_Float16)d.y; r[14] = (_Float16)d.z; r[15] = (_Float16)d.w;
  return r;
}

__device__ __forceinline__ v16h frag_a32(const float* arow, int k0) {
  int half = (threadIdx.x >> 4) & 1;
  const float4* p = (const float4*)(arow + k0 + (half << 3));
  const float4* q = (const float4*)(arow + k0 + 16 + (half << 3));
  return cvt16(p[0], p[1], q[0], q[1]);
}

__device__ __forceinline__ v16h frag_b32(const float* bcol, int k0) {
  int half = (threadIdx.x >> 4) & 1;
  const float4* p = (const float4*)(bcol + k0 + (half << 3));
  const float4* q = (const float4*)(bcol + k0 + 16 + (half << 3));
  return cvt16(p[0], p[1], q[0], q[1]);
}

__device__ __forceinline__ v8f wmma_f16(v16h a, v16h b, v8f c) {
  return __builtin_amdgcn_wmma_f32_16x16x32_f16(false, a, false, b, (short)0, c,
                                                false, false);
}

template <int K>
__device__ __forceinline__ v8f gemm_acc_t(const float* arow, const float* bcol,
                                          v8f acc) {
#pragma unroll
  for (int k0 = 0; k0 < K; k0 += 32)
    acc = wmma_f16(frag_a32(arow, k0), frag_b32(bcol, k0), acc);
  return acc;
}

// one A fragment feeds two adjacent N tiles (2x reuse, 2 WMMAs per K step)
template <int K>
__device__ __forceinline__ void gemm_acc2(const float* arow, const float* b0,
                                          const float* b1, v8f& a0, v8f& a1) {
#pragma unroll
  for (int k0 = 0; k0 < K; k0 += 32) {
    v16h af = frag_a32(arow, k0);
    a0 = wmma_f16(af, frag_b32(b0, k0), a0);
    a1 = wmma_f16(af, frag_b32(b1, k0), a1);
  }
}

// ---------------------------------------------------------------------------
// small utility kernels
// ---------------------------------------------------------------------------
__global__ void k_zero(float* __restrict__ p, int n) {
  int i = blockIdx.x * 256 + threadIdx.x;
  if (i < n) p[i] = 0.f;
}

__global__ void k_nchw2nlc(const float* __restrict__ src, float* __restrict__ dst,
                           size_t n, int C, int L) {
  size_t idx = (size_t)blockIdx.x * 256 + threadIdx.x;
  if (idx >= n) return;
  int l = (int)(idx % L);
  int c = (int)((idx / L) % C);
  int b = (int)(idx / ((size_t)L * C));
  dst[((size_t)b * L + l) * C + c] = src[idx];
}

// LayerNorm over last dim (one wave per row, C <= 512)
__global__ void k_ln(const float* __restrict__ x, const float* __restrict__ w,
                     const float* __restrict__ bb, float* __restrict__ y,
                     int R, int C) {
  int row = blockIdx.x * (blockDim.x >> 5) + (threadIdx.x >> 5);
  if (row >= R) return;
  int lane = threadIdx.x & 31;
  const float* xr = x + (size_t)row * C;
  float v[16];
  int cnt = 0;
  float s = 0.f;
  for (int c = lane; c < C; c += 32) { float t = xr[c]; v[cnt++] = t; s += t; }
  for (int m = 1; m < 32; m <<= 1) s += __shfl_xor(s, m, 32);
  float mean = s / (float)C;
  float q = 0.f;
  for (int i = 0; i < cnt; ++i) { float d = v[i] - mean; q += d * d; }
  for (int m = 1; m < 32; m <<= 1) q += __shfl_xor(q, m, 32);
  float rinv = rsqrtf(q / (float)C + 1e-5f);
  float* yr = y + (size_t)row * C;
  int i = 0;
  for (int c = lane; c < C; c += 32) yr[c] = (v[i++] - mean) * rinv * w[c] + bb[c];
}

// weight repacks for implicit-GEMM convolutions (k-contiguous B layout)
__global__ void k_wt_fore(const float* __restrict__ w, float* __restrict__ wt) {
  int idx = blockIdx.x * 256 + threadIdx.x;          // [tap][co][ci], 9*128*64
  if (idx >= 9 * 128 * 64) return;
  int tap = idx / (128 * 64);
  int rem = idx - tap * (128 * 64);
  int co = rem >> 6, ci = rem & 63;
  int dy = tap / 3, dx = tap % 3;
  wt[idx] = w[((co * 64 + ci) * 3 + dy) * 3 + dx];   // fore_k_w [128,64,3,3]
}

__global__ void k_wt_post(const float* __restrict__ w, float* __restrict__ wt) {
  int idx = blockIdx.x * 256 + threadIdx.x;          // [par][co][ci], 4*128*256
  if (idx >= 4 * 128 * 256) return;
  int par = idx / (128 * 256);
  int rem = idx - par * (128 * 256);
  int co = rem >> 8, ci = rem & 255;
  int py = par >> 1, px = par & 1;
  wt[idx] = w[((ci * 128 + co) * 2 + py) * 2 + px];  // post_k_w [256,128,2,2]
}

// ---------------------------------------------------------------------------
// fore K-projection: Conv2d(64->128, k3, s2, p1), channel-last implicit GEMM.
// 9 taps x K=64; border rows read a zeroed dummy row (branchless loads).
// tile = (b<8>, mtile<64>, ntile2<4>); dual 16-wide N tiles per wave.
// ---------------------------------------------------------------------------
__global__ void __launch_bounds__(128) k_conv_fore(const float* __restrict__ nf,
                                                   const float* __restrict__ wt,
                                                   const float* __restrict__ bias,
                                                   const float* __restrict__ zrow,
                                                   float* __restrict__ out) {
  int gw = blockIdx.x * 4 + (threadIdx.x >> 5);   // 2048 wave-tiles
  int nt = gw & 3, mt = (gw >> 2) & 63, b = gw >> 8;
  int lane = threadIdx.x & 31, mn = lane & 15, half = lane >> 4;
  int m = mt * 16 + mn, oi = m >> 5, oj = m & 31;
  const float* b0 = wt + (size_t)(nt * 32 + mn) * 64;
  const float* b1 = b0 + (size_t)16 * 64;
  v8f a0 = {}, a1 = {};
#pragma unroll
  for (int dy = 0; dy < 3; ++dy)
#pragma unroll
    for (int dx = 0; dx < 3; ++dx) {
      int r = 2 * oi - 1 + dy, c = 2 * oj - 1 + dx;
      bool ok = (r >= 0 && r < 64 && c >= 0 && c < 64);
      const float* arow = ok ? nf + ((size_t)b * 4096 + r * 64 + c) * 64 : zrow;
      size_t toff = (size_t)(dy * 3 + dx) * (128 * 64);
      gemm_acc2<64>(arow, b0 + toff, b1 + toff, a0, a1);
    }
#pragma unroll
  for (int i = 0; i < 8; ++i) {
    int row = mt * 16 + half * 8 + i;
    int c0 = nt * 32 + mn, c1 = c0 + 16;
    out[((size_t)b * NL + row) * NC + c0] = a0[i] + bias[c0];
    out[((size_t)b * NL + row) * NC + c1] = a1[i] + bias[c1];
  }
}

// ---------------------------------------------------------------------------
// post K-projection: ConvTranspose2d(256->128, k2, s2) = 4 parity GEMMs K=256
// tile = (b<8>, par<4>, mtile<16>, ntile2<4>); scattered stores.
// ---------------------------------------------------------------------------
__global__ void __launch_bounds__(128) k_conv_post(const float* __restrict__ np,
                                                   const float* __restrict__ wt,
                                                   const float* __restrict__ bias,
                                                   float* __restrict__ out) {
  int gw = blockIdx.x * 4 + (threadIdx.x >> 5);   // 2048 wave-tiles
  int nt = gw & 3, mt = (gw >> 2) & 15, par = (gw >> 6) & 3, b = gw >> 8;
  int lane = threadIdx.x & 31, mn = lane & 15, half = lane >> 4;
  const float* arow = np + ((size_t)b * 256 + mt * 16 + mn) * 256;
  const float* b0 = wt + (size_t)(par * 128 + nt * 32 + mn) * 256;
  const float* b1 = b0 + (size_t)16 * 256;
  v8f a0 = {}, a1 = {};
  gemm_acc2<256>(arow, b0, b1, a0, a1);
  int py = par >> 1, px = par & 1;
#pragma unroll
  for (int i = 0; i < 8; ++i) {
    int mrow = mt * 16 + half * 8 + i;
    int ii = mrow >> 4, jj = mrow & 15;
    int pos = (2 * ii + py) * NW + (2 * jj + px);
    int c0 = nt * 32 + mn, c1 = c0 + 16;
    out[((size_t)b * NL + pos) * NC + c0] = a0[i] + bias[c0];
    out[((size_t)b * NL + pos) * NC + c1] = a1[i] + bias[c1];
  }
}

// ---------------------------------------------------------------------------
// Generic WMMA GEMM, dual N tile: C = A[M,K] @ W[N,K]^T + bias (+res)
// MODE 0: plain; 1: +residual; 2: +residual, transposed store to [B,N,Lout]
// ---------------------------------------------------------------------------
template <int K, int MODE>
__global__ void __launch_bounds__(128) k_gemm2(const float* __restrict__ A,
                                               const float* __restrict__ W,
                                               const float* __restrict__ bias,
                                               const float* __restrict__ R,
                                               float* __restrict__ C,
                                               int M, int N, int Lout) {
  int gw = blockIdx.x * 4 + (threadIdx.x >> 5);
  int nt2 = N >> 5;
  int total = (M >> 4) * nt2;
  if (gw >= total) return;
  int mt = gw / nt2, nt = gw - mt * nt2;
  int lane = threadIdx.x & 31, mn = lane & 15, half = lane >> 4;
  const float* arow = A + (size_t)(mt * 16 + mn) * K;
  const float* b0 = W + (size_t)(nt * 32 + mn) * K;
  const float* b1 = b0 + (size_t)16 * K;
  v8f a0 = {}, a1 = {};
  gemm_acc2<K>(arow, b0, b1, a0, a1);
#pragma unroll
  for (int i = 0; i < 8; ++i) {
    int row = mt * 16 + half * 8 + i;
    int c0 = nt * 32 + mn, c1 = c0 + 16;
    float v0 = a0[i] + bias[c0];
    float v1 = a1[i] + bias[c1];
    if (MODE >= 1) {
      v0 += R[(size_t)row * N + c0];
      v1 += R[(size_t)row * N + c1];
    }
    if (MODE == 2) {
      int b = row / Lout, pos = row - b * Lout;
      C[((size_t)b * N + c0) * Lout + pos] = v0;
      C[((size_t)b * N + c1) * Lout + pos] = v1;
    } else {
      C[(size_t)row * N + c0] = v0;
      C[(size_t)row * N + c1] = v1;
    }
  }
}

// ---------------------------------------------------------------------------
// QV projection (K=128, N=256), dual N tile, split store q/v [B,L,128]
// ---------------------------------------------------------------------------
__global__ void __launch_bounds__(128) k_qv(const float* __restrict__ A,
                                            const float* __restrict__ W,
                                            const float* __restrict__ bias,
                                            float* __restrict__ qo,
                                            float* __restrict__ vo, int M) {
  int gw = blockIdx.x * 4 + (threadIdx.x >> 5);
  int total = (M >> 4) * 8;
  if (gw >= total) return;
  int mt = gw >> 3, nt = gw & 7;
  int lane = threadIdx.x & 31, mn = lane & 15, half = lane >> 4;
  const float* arow = A + (size_t)(mt * 16 + mn) * 128;
  const float* b0 = W + (size_t)(nt * 32 + mn) * 128;
  const float* b1 = b0 + (size_t)16 * 128;
  v8f a0 = {}, a1 = {};
  gemm_acc2<128>(arow, b0, b1, a0, a1);
#pragma unroll
  for (int i = 0; i < 8; ++i) {
    int row = mt * 16 + half * 8 + i;
    int c0 = nt * 32 + mn, c1 = c0 + 16;
    float v0 = a0[i] + bias[c0];
    float v1 = a1[i] + bias[c1];
    if (c0 < 128) qo[(size_t)row * 128 + c0] = v0;
    else          vo[(size_t)row * 128 + (c0 - 128)] = v0;
    if (c1 < 128) qo[(size_t)row * 128 + c1] = v1;
    else          vo[(size_t)row * 128 + (c1 - 128)] = v1;
  }
}

// ---------------------------------------------------------------------------
// Flash-style cross attention per wave: 16 queries x 64 key tiles.
// q,k,v,o all [B, L, C] channel-last; head h uses cols h*16..h*16+15.
// Branchless tile loads; P relayout through LDS (stride-20 rows, b128 reads).
// ---------------------------------------------------------------------------
__global__ void __launch_bounds__(128) k_attn(const float* __restrict__ Q,
                                              const float* __restrict__ Kb,
                                              const float* __restrict__ V,
                                              float* __restrict__ O) {
  __shared__ float lds[4][16 * 20];
  int wid = threadIdx.x >> 5;
  int gw = blockIdx.x * 4 + wid;
  int qt = gw & 63, h = (gw >> 6) & 7, b = gw >> 9;
  int lane = threadIdx.x & 31, mn = lane & 15, half = lane >> 4;
  float msk = half ? 0.f : 1.f;   // upper K-half of padded fragments is zero

  // Q fragment (A layout, K=16 zero-padded to 32), pre-scaled by hd^-0.5
  const float* qrow = Q + ((size_t)(b * NL + qt * 16 + mn)) * NC + h * NHD;
  v16h aq;
  {
    const float4* p = (const float4*)(qrow + (half << 3));
    float4 x = p[0], y = p[1];
    aq[0] = (_Float16)(x.x * 0.25f); aq[1] = (_Float16)(x.y * 0.25f);
    aq[2] = (_Float16)(x.z * 0.25f); aq[3] = (_Float16)(x.w * 0.25f);
    aq[4] = (_Float16)(y.x * 0.25f); aq[5] = (_Float16)(y.y * 0.25f);
    aq[6] = (_Float16)(y.z * 0.25f); aq[7] = (_Float16)(y.w * 0.25f);
#pragma unroll
    for (int i = 8; i < 16; ++i) aq[i] = (_Float16)0.f;
  }

  float rm[8], rs[8];
#pragma unroll
  for (int i = 0; i < 8; ++i) { rm[i] = -1e30f; rs[i] = 0.f; }
  v8f oacc = {};

  for (int kt = 0; kt < 64; ++kt) {
    // K tile (B layout): col n = key kt*16+mn; contraction = head dim (16->32)
    const float* krow = Kb + ((size_t)(b * NL + kt * 16 + mn)) * NC + h * NHD;
    const float4* kp = (const float4*)(krow + (half << 3));
    float4 k0 = kp[0], k1 = kp[1];
    float4 k2 = make_float4(0.f, 0.f, 0.f, 0.f), k3 = k2;
    v16h bk = cvt16(k0, k1, k2, k3);

    v8f zero = {};
    v8f s = wmma_f16(aq, bk, zero);

    // online softmax per row (rows = half*8+i, cols across the 16-lane half)
#pragma unroll
    for (int i = 0; i < 8; ++i) {
      float x = s[i];
      float tm = x;
      tm = fmaxf(tm, __shfl_xor(tm, 1, 32));
      tm = fmaxf(tm, __shfl_xor(tm, 2, 32));
      tm = fmaxf(tm, __shfl_xor(tm, 4, 32));
      tm = fmaxf(tm, __shfl_xor(tm, 8, 32));
      float nm = fmaxf(rm[i], tm);
      float pe = __expf(x - nm);
      float cr = __expf(rm[i] - nm);
      float ps = pe;
      ps += __shfl_xor(ps, 1, 32);
      ps += __shfl_xor(ps, 2, 32);
      ps += __shfl_xor(ps, 4, 32);
      ps += __shfl_xor(ps, 8, 32);
      rs[i] = rs[i] * cr + ps;
      rm[i] = nm;
      oacc[i] *= cr;
      lds[wid][(half * 8 + i) * 20 + mn] = pe;   // stage P tile for relayout
    }
    asm volatile("s_wait_dscnt 0" ::: "memory");

    // P tile re-read in A layout (row = mn, contraction = key idx, 16->32)
    v16h ap;
    {
      const float4* lp = (const float4*)&lds[wid][mn * 20 + (half << 3)];
      float4 x = lp[0], y = lp[1];
      ap[0] = (_Float16)x.x; ap[1] = (_Float16)x.y;
      ap[2] = (_Float16)x.z; ap[3] = (_Float16)x.w;
      ap[4] = (_Float16)y.x; ap[5] = (_Float16)y.y;
      ap[6] = (_Float16)y.z; ap[7] = (_Float16)y.w;
#pragma unroll
      for (int i = 8; i < 16; ++i) ap[i] = (_Float16)0.f;
    }
    asm volatile("s_wait_dscnt 0" ::: "memory");

    // V tile (B layout): col n = head dim mn; contraction = key idx (16->32)
    v16h bv;
#pragma unroll
    for (int i = 0; i < 8; ++i) {
      int kk = (half << 3) + i;
      float x0 = V[((size_t)(b * NL + kt * 16 + kk)) * NC + h * NHD + mn];
      bv[i] = (_Float16)x0; bv[8 + i] = (_Float16)0.f;
    }
    oacc = wmma_f16(ap, bv, oacc);
  }

#pragma unroll
  for (int i = 0; i < 8; ++i) {
    int row = qt * 16 + half * 8 + i;
    O[((size_t)(b * NL + row)) * NC + h * NHD + mn] = oacc[i] / rs[i];
  }
}

// ---------------------------------------------------------------------------
// Depthwise 3x3 (pad 1) over [B,32,32,512] channel-last; out = dw+bias(+x)
// ---------------------------------------------------------------------------
__global__ void k_dwconv(const float* __restrict__ x, const float* __restrict__ w,
                         const float* __restrict__ bias, float* __restrict__ y,
                         int skip) {
  size_t idx = (size_t)blockIdx.x * 256 + threadIdx.x;
  if (idx >= (size_t)NB * NL * NHID) return;
  int c = (int)(idx & (NHID - 1));
  size_t bp = idx >> 9;
  int pos = (int)(bp & (NL - 1));
  int b = (int)(bp >> 10);
  int i = pos >> 5, j = pos & 31;
  float acc = bias[c];
#pragma unroll
  for (int dy = 0; dy < 3; ++dy)
#pragma unroll
    for (int dx = 0; dx < 3; ++dx) {
      int r = i + dy - 1, cc = j + dx - 1;
      if (r >= 0 && r < NH && cc >= 0 && cc < NW)
        acc += x[(((size_t)b * NL + r * NW + cc) * NHID) + c] * w[c * 9 + dy * 3 + dx];
    }
  if (skip) acc += x[idx];
  y[idx] = acc;
}

__global__ void k_gelu(float* __restrict__ x, size_t n) {
  size_t i = (size_t)blockIdx.x * 256 + threadIdx.x;
  if (i < n) {
    float v = x[i];
    x[i] = 0.5f * v * (1.f + erff(v * 0.70710678118654752f));
  }
}

// ---------------------------------------------------------------------------
// Host orchestration
// ---------------------------------------------------------------------------
extern "C" void kernel_launch(void* const* d_in, const int* in_sizes, int n_in,
                              void* d_out, int out_size, void* d_ws, size_t ws_size,
                              hipStream_t stream) {
  (void)in_sizes; (void)n_in; (void)out_size; (void)ws_size;
  const float* x      = (const float*)d_in[0];
  const float* fore_x = (const float*)d_in[1];
  const float* post_x = (const float*)d_in[2];
  auto P = [&](int i) { return (const float*)d_in[i]; };
  const float *ln_x1_w = P(3),  *ln_x1_b = P(4);
  const float *ln_fore_w = P(5), *ln_fore_b = P(6);
  const float *fore_qv_w = P(7), *fore_qv_b = P(8);
  const float *fore_k_w = P(9),  *fore_k_b = P(10);
  const float *fore_proj_w = P(11), *fore_proj_b = P(12);
  const float *ln_x2_w = P(13), *ln_x2_b = P(14);
  const float *m1_fc1_w = P(15), *m1_fc1_b = P(16);
  const float *m1_dw_w = P(17),  *m1_dw_b = P(18);
  const float *m1_fc2_w = P(19), *m1_fc2_b = P(20);
  const float *ln_x3_w = P(21), *ln_x3_b = P(22);
  const float *ln_post_w = P(23), *ln_post_b = P(24);
  const float *post_qv_w = P(25), *post_qv_b = P(26);
  const float *post_k_w = P(27),  *post_k_b = P(28);
  const float *post_proj_w = P(29), *post_proj_b = P(30);
  const float *ln_x4_w = P(31), *ln_x4_b = P(32);
  const float *m2_fc1_w = P(33), *m2_fc1_b = P(34);
  const float *m2_dw_w = P(35),  *m2_dw_b = P(36);
  const float *m2_ln_w = P(37),  *m2_ln_b = P(38);
  const float *m2_fc2_w = P(39), *m2_fc2_b = P(40);

  // workspace layout (floats)
  float* wsf = (float*)d_ws;
  size_t off = 0;
  auto alloc = [&](size_t n) { float* p = wsf + off; off += n; return p; };
  const size_t SLC = (size_t)NB * NL * NC;        // 1M floats
  float* xt0   = alloc(SLC);
  float* add1  = alloc(SLC);
  float* xt1   = alloc(SLC);
  float* add3  = alloc(SLC);
  float* nx    = alloc(SLC);
  float* foret = alloc((size_t)NB * 4096 * 64);   // 2M
  float* postt = alloc((size_t)NB * 256 * 256);   // 0.5M
  float* kbuf  = alloc(SLC);
  float* qb    = alloc(SLC);
  float* vb    = alloc(SLC);
  float* ob    = alloc(SLC);
  float* h1    = alloc((size_t)NB * NL * NHID);   // 4M
  float* dwb   = alloc((size_t)NB * NL * NHID);   // 4M
  float* zrow  = alloc(64);                       // zero pad row for conv
  float* wtf   = alloc(9 * 128 * 64);
  float* wtp   = alloc(4 * 128 * 256);

  const int M = NB * NL;                          // 8192 token rows
  dim3 blk(128);

  // ---- stage 0: transposes + weight repacks + zero row ----
  k_zero<<<1, 256, 0, stream>>>(zrow, 64);
  k_nchw2nlc<<<(int)((SLC + 255) / 256), 256, 0, stream>>>(x, xt0, SLC, NC, NL);
  k_nchw2nlc<<<(int)(((size_t)NB * 4096 * 64 + 255) / 256), 256, 0, stream>>>(
      fore_x, foret, (size_t)NB * 4096 * 64, 64, 4096);
  k_nchw2nlc<<<(int)(((size_t)NB * 256 * 256 + 255) / 256), 256, 0, stream>>>(
      post_x, postt, (size_t)NB * 256 * 256, 256, 256);
  k_wt_fore<<<(9 * 128 * 64 + 255) / 256, 256, 0, stream>>>(fore_k_w, wtf);
  k_wt_post<<<(4 * 128 * 256 + 255) / 256, 256, 0, stream>>>(post_k_w, wtp);

  // ---- stage 1: fore cross-attention ----
  k_ln<<<M / 4, blk, 0, stream>>>(xt0, ln_x1_w, ln_x1_b, nx, M, NC);
  k_ln<<<(NB * 4096) / 4, blk, 0, stream>>>(foret, ln_fore_w, ln_fore_b, foret,
                                            NB * 4096, 64);
  k_conv_fore<<<512, blk, 0, stream>>>(foret, wtf, fore_k_b, zrow, kbuf);
  k_qv<<<((M / 16) * 8) / 4, blk, 0, stream>>>(nx, fore_qv_w, fore_qv_b, qb, vb, M);
  k_attn<<<(NB * NHEADS * (NL / 16)) / 4, blk, 0, stream>>>(qb, kbuf, vb, ob);
  k_gemm2<128, 1><<<((M / 16) * (NC / 32)) / 4, blk, 0, stream>>>(
      ob, fore_proj_w, fore_proj_b, xt0, add1, M, NC, 0);

  // ---- MixFFN 1 (skip=False) ----
  k_ln<<<M / 4, blk, 0, stream>>>(add1, ln_x2_w, ln_x2_b, nx, M, NC);
  k_gemm2<128, 0><<<((M / 16) * (NHID / 32)) / 4, blk, 0, stream>>>(
      nx, m1_fc1_w, m1_fc1_b, nullptr, h1, M, NHID, 0);
  k_dwconv<<<(int)(((size_t)M * NHID + 255) / 256), 256, 0, stream>>>(
      h1, m1_dw_w, m1_dw_b, dwb, 0);
  k_gelu<<<(int)(((size_t)M * NHID + 255) / 256), 256, 0, stream>>>(
      dwb, (size_t)M * NHID);
  k_gemm2<512, 1><<<((M / 16) * (NC / 32)) / 4, blk, 0, stream>>>(
      dwb, m1_fc2_w, m1_fc2_b, add1, xt1, M, NC, 0);

  // ---- stage 2: post cross-attention ----
  k_ln<<<M / 4, blk, 0, stream>>>(xt1, ln_x3_w, ln_x3_b, nx, M, NC);
  k_ln<<<(NB * 256) / 4, blk, 0, stream>>>(postt, ln_post_w, ln_post_b, postt,
                                           NB * 256, 256);
  k_conv_post<<<512, blk, 0, stream>>>(postt, wtp, post_k_b, kbuf);
  k_qv<<<((M / 16) * 8) / 4, blk, 0, stream>>>(nx, post_qv_w, post_qv_b, qb, vb, M);
  k_attn<<<(NB * NHEADS * (NL / 16)) / 4, blk, 0, stream>>>(qb, kbuf, vb, ob);
  k_gemm2<128, 1><<<((M / 16) * (NC / 32)) / 4, blk, 0, stream>>>(
      ob, post_proj_w, post_proj_b, xt1, add3, M, NC, 0);

  // ---- MixFFN 2 (skip=True, inner LN) ----
  k_ln<<<M / 4, blk, 0, stream>>>(add3, ln_x4_w, ln_x4_b, nx, M, NC);
  k_gemm2<128, 0><<<((M / 16) * (NHID / 32)) / 4, blk, 0, stream>>>(
      nx, m2_fc1_w, m2_fc1_b, nullptr, h1, M, NHID, 0);
  k_dwconv<<<(int)(((size_t)M * NHID + 255) / 256), 256, 0, stream>>>(
      h1, m2_dw_w, m2_dw_b, dwb, 1);
  k_ln<<<M / 4, blk, 0, stream>>>(dwb, m2_ln_w, m2_ln_b, h1, M, NHID);
  k_gelu<<<(int)(((size_t)M * NHID + 255) / 256), 256, 0, stream>>>(
      h1, (size_t)M * NHID);
  // final fc2 + residual, fused transposed store into d_out [B,C,H,W]
  k_gemm2<512, 2><<<((M / 16) * (NC / 32)) / 4, blk, 0, stream>>>(
      h1, m2_fc2_w, m2_fc2_b, add3, (float*)d_out, M, NC, NL);
}
